// GCPNetDecoder_25340307046878
// MI455X (gfx1250) — matrix operand order repaired and checked by reading, not compiled, not run on hardware
//
#include <hip/hip_runtime.h>
#include <hip/hip_bf16.h>

// ---------------- problem constants ----------------
#define B_      16
#define L_      512
#define D_      128
#define K_      16
#define NRBF    16
#define NLAYERS 4
#define N_      (B_ * L_)          // 8192 nodes
#define KE      272                // 2*D + NRBF  (edge feature dim)
#define KEP     288                // padded to 9 chunks of 32
#define KH      258                // 2*D + 2     (node feature dim)
#define KHP     288
#define NCHUNK  9                  // KEP/32
#define NTILE   8                  // 128/16 column tiles
#define FRAG    512                // halfs per packed 32x16 B fragment

typedef __attribute__((ext_vector_type(16))) _Float16 v16h;
typedef __attribute__((ext_vector_type(8)))  float    v8f;
typedef __attribute__((ext_vector_type(4)))  unsigned int u32x4;
typedef __attribute__((ext_vector_type(4)))  int v4i;

// ---------------- async Global->LDS copy (CDNA5 path, VGPR-bypassing) ----------------
#if defined(__has_builtin)
#  if __has_builtin(__builtin_amdgcn_global_load_async_to_lds_b128)
#    define HAVE_ASYNC_LDS 1
#  endif
#endif
#ifndef HAVE_ASYNC_LDS
#  define HAVE_ASYNC_LDS 0
#endif

__device__ __forceinline__ void async_copy_b128(const void* g, void* l) {
#if HAVE_ASYNC_LDS
    void* gnc = (void*)g;   // builtin takes non-const global pointer
    __builtin_amdgcn_global_load_async_to_lds_b128(
        (__attribute__((address_space(1))) v4i*)gnc,
        (__attribute__((address_space(3))) v4i*)l, 0, 0);
#else
    *(u32x4*)l = *(const u32x4*)g;
#endif
}

__device__ __forceinline__ void async_wait_all() {
#if HAVE_ASYNC_LDS
#  if defined(__has_builtin) && __has_builtin(__builtin_amdgcn_s_wait_asynccnt)
    __builtin_amdgcn_s_wait_asynccnt(0);
#  else
    asm volatile("s_wait_asynccnt 0x0" ::: "memory");
#  endif
#endif
}

// ================= weight packing into WMMA B-fragment layout =================
// B (32x16, f16, wave32): lane<16 holds K=0..7,16..23 (elems j), lane>=16 K=8..15,24..31,
// column n = lane&15.  Packed flat: [((l*9+kc)*8+t)*512 + lane*16 + j]
__global__ void pack_we_kernel(const float* __restrict__ We, _Float16* __restrict__ WeP) {
    int id = blockIdx.x * blockDim.x + threadIdx.x;
    if (id >= NLAYERS * NCHUNK * NTILE * FRAG) return;
    int e    = id & 511;
    int t    = (id >> 9) & 7;
    int r2   = id >> 12;           // l*9 + kc
    int kc   = r2 % NCHUNK;
    int l    = r2 / NCHUNK;
    int lane = e >> 4;
    int j    = e & 15;
    int n    = (lane & 15) + t * 16;
    int k    = kc * 32 + ((lane >= 16) ? 8 : 0) + j + ((j >= 8) ? 8 : 0);
    float v  = (k < KE) ? We[(l * KE + k) * D_ + n] : 0.0f;
    WeP[id] = (_Float16)v;
}

__global__ void pack_wh_kernel(const float* __restrict__ Wh, _Float16* __restrict__ WhP) {
    int id = blockIdx.x * blockDim.x + threadIdx.x;
    if (id >= NLAYERS * NCHUNK * NTILE * FRAG) return;
    int e    = id & 511;
    int t    = (id >> 9) & 7;
    int r2   = id >> 12;
    int kc   = r2 % NCHUNK;
    int l    = r2 / NCHUNK;
    int lane = e >> 4;
    int j    = e & 15;
    int n    = (lane & 15) + t * 16;
    int k    = kc * 32 + ((lane >= 16) ? 8 : 0) + j + ((j >= 8) ? 8 : 0);
    float v  = (k < KH) ? Wh[(l * KH + k) * D_ + n] : 0.0f;
    WhP[id] = (_Float16)v;
}

// ================= h initialization (f16 working copy) =================
__global__ void seth_kernel(const float* __restrict__ x, _Float16* __restrict__ h16) {
    int id = blockIdx.x * blockDim.x + threadIdx.x;
    if (id >= N_ * D_) return;
    h16[id] = (_Float16)x[id];
}

// ================= x_bb = h @ W_init  (128 -> 9) =================
__global__ void init_xbb_kernel(const float* __restrict__ x, const float* __restrict__ Winit,
                                float* __restrict__ xbb) {
    int id = blockIdx.x * blockDim.x + threadIdx.x;
    if (id >= N_ * 9) return;
    int n = id / 9, j = id - 9 * n;
    const float* xr = x + (size_t)n * D_;
    float s = 0.0f;
    #pragma unroll 8
    for (int f = 0; f < D_; ++f) s += xr[f] * Winit[f * 9 + j];
    xbb[id] = s;
}

// ================= recenter: mean of CA (atom 1) over all nodes =================
__global__ void center_reduce_kernel(const float* __restrict__ xbb, float* __restrict__ mean) {
    __shared__ float sx[256], sy[256], sz[256];
    int tid = threadIdx.x;
    float ax = 0.f, ay = 0.f, az = 0.f;
    for (int n = tid; n < N_; n += 256) {
        ax += xbb[n * 9 + 3];
        ay += xbb[n * 9 + 4];
        az += xbb[n * 9 + 5];
    }
    sx[tid] = ax; sy[tid] = ay; sz[tid] = az;
    __syncthreads();
    for (int s = 128; s > 0; s >>= 1) {
        if (tid < s) { sx[tid] += sx[tid + s]; sy[tid] += sy[tid + s]; sz[tid] += sz[tid + s]; }
        __syncthreads();
    }
    if (tid == 0) {
        float inv = 1.0f / (float)N_;
        mean[0] = sx[0] * inv; mean[1] = sy[0] * inv; mean[2] = sz[0] * inv;
    }
}

__global__ void center_apply_kernel(float* __restrict__ xbb, const float* __restrict__ mean) {
    int id = blockIdx.x * blockDim.x + threadIdx.x;
    if (id >= N_ * 9) return;
    xbb[id] -= mean[id % 3];
}

// ================= per-chain kNN + edge geometry + chi norms =================
__global__ void knn_kernel(const float* __restrict__ xbb, int* __restrict__ idx,
                           float* __restrict__ dist, float* __restrict__ xi,
                           float* __restrict__ chin) {
    int n = blockIdx.x * blockDim.x + threadIdx.x;
    if (n >= N_) return;
    int b = n >> 9;                 // n / L_
    int l = n & (L_ - 1);
    float qx = xbb[n * 9 + 3], qy = xbb[n * 9 + 4], qz = xbb[n * 9 + 5];

    float bd[K_]; int bi[K_];
    #pragma unroll
    for (int k = 0; k < K_; ++k) { bd[k] = 3.4e38f; bi[k] = n; }

    int nbase = b * L_;
    for (int j = 0; j < L_; ++j) {
        if (j == l) continue;
        int m = nbase + j;
        float dx = xbb[m * 9 + 3] - qx;
        float dy = xbb[m * 9 + 4] - qy;
        float dz = xbb[m * 9 + 5] - qz;
        float d2 = dx * dx + dy * dy + dz * dz;
        if (d2 < bd[K_ - 1]) {
            int pos = 0;
            #pragma unroll
            for (int k = 0; k < K_ - 1; ++k) pos += (bd[k] <= d2) ? 1 : 0;
            #pragma unroll
            for (int k = K_ - 1; k > 0; --k) {
                bool mv = (k > pos);
                bd[k] = mv ? bd[k - 1] : bd[k];
                bi[k] = mv ? bi[k - 1] : bi[k];
            }
            #pragma unroll
            for (int k = 0; k < K_; ++k) {
                if (k == pos) { bd[k] = d2; bi[k] = m; }
            }
        }
    }

    #pragma unroll
    for (int k = 0; k < K_; ++k) {
        int m = bi[k];
        float dx = xbb[m * 9 + 3] - qx;
        float dy = xbb[m * 9 + 4] - qy;
        float dz = xbb[m * 9 + 5] - qz;
        float d  = sqrtf(dx * dx + dy * dy + dz * dz);
        int eo = n * K_ + k;
        idx[eo]  = m;
        dist[eo] = d;
        float inv = 1.0f / (d + 1e-8f);
        xi[eo * 3 + 0] = dx * inv;
        xi[eo * 3 + 1] = dy * inv;
        xi[eo * 3 + 2] = dz * inv;
    }

    // chi norms: ||v||/(||v||+eps) for fwd/bwd chain vectors (0 at ends)
    float fn = 0.f, bn = 0.f;
    if (l < L_ - 1) {
        float vx = xbb[(n + 1) * 9 + 3] - qx;
        float vy = xbb[(n + 1) * 9 + 4] - qy;
        float vz = xbb[(n + 1) * 9 + 5] - qz;
        float nv = sqrtf(vx * vx + vy * vy + vz * vz);
        fn = nv / (nv + 1e-8f);
    }
    if (l > 0) {
        float vx = xbb[(n - 1) * 9 + 3] - qx;
        float vy = xbb[(n - 1) * 9 + 4] - qy;
        float vz = xbb[(n - 1) * 9 + 5] - qz;
        float nv = sqrtf(vx * vx + vy * vy + vz * vz);
        bn = nv / (nv + 1e-8f);
    }
    chin[n * 2 + 0] = fn;
    chin[n * 2 + 1] = bn;
}

// ================= edge MLP (WMMA) + aggregation + gate + dx =================
// One workgroup per node: M-tile = the node's 16 edges, 8 waves cover N=128.
__global__ __launch_bounds__(256) void msg_kernel(
    const _Float16* __restrict__ h16, const int* __restrict__ idx,
    const float* __restrict__ dist, const float* __restrict__ xi,
    const _Float16* __restrict__ WeP, const float* __restrict__ Wg,
    _Float16* __restrict__ hagg16, float* __restrict__ xbb, int layer) {

    const int node = blockIdx.x;
    const int tid  = threadIdx.x;
    const int lane = tid & 31;
    const int wave = tid >> 5;

    __shared__ _Float16 eL[16 * KEP];      // 16 x 288 edge-feature tile (f16)
    __shared__ float    msgL[16 * D_];     // 16 x 128 relu(msg)
    __shared__ float    gateL[16 * 3];

    // --- assemble e = [h_src | h_dst | rbf(dist)] in LDS ---
    // Gathered h rows go Global->LDS via async b128 (no VGPR staging);
    // the 32 rbf/pad columns per row are computed in VALU meanwhile.
    {
        int k = tid >> 4;                  // edge row
        int part = tid & 15;               // 16-byte segment within a 128-half row
        int src = idx[node * K_ + k];
        async_copy_b128(h16 + (size_t)src  * D_ + part * 8, &eL[k * KEP + part * 8]);
        async_copy_b128(h16 + (size_t)node * D_ + part * 8, &eL[k * KEP + 128 + part * 8]);
        float dk = dist[node * K_ + k];
        #pragma unroll
        for (int s = 0; s < 2; ++s) {
            int c = 256 + part * 2 + s;
            _Float16 v = (_Float16)0.0f;
            if (c < 272) {
                int j = c - 256;
                float mu = (20.0f / 15.0f) * (float)j;
                float t = (dk - mu) * 0.8f;          // sigma = 1.25
                v = (_Float16)__expf(-t * t);
            }
            eL[k * KEP + c] = v;
        }
    }
    async_wait_all();
    __syncthreads();

    // --- 16x128 GEMM over K=288 via 9x v_wmma_f32_16x16x32_f16 ---
    v8f c = {};
    const int m    = lane & 15;
    const int hoff = (lane >= 16) ? 8 : 0;
    #pragma unroll
    for (int kc = 0; kc < NCHUNK; ++kc) {
        v16h a, bfr;
        const _Float16* ar = &eL[m * KEP + kc * 32 + hoff];
        #pragma unroll
        for (int j = 0; j < 16; ++j) a[j] = ar[j + ((j >= 8) ? 8 : 0)];
        const _Float16* br = WeP + ((size_t)((layer * NCHUNK + kc) * NTILE + wave)) * FRAG + lane * 16;
        bfr = *(const v16h*)br;
        c = __builtin_amdgcn_wmma_f32_16x16x32_f16(false, a, false, bfr, (short)0, c, false, false);
    }

    // --- relu, spill msg tile to LDS (C layout: row = r + 8*(lane>=16), col = lane&15) ---
    {
        int n   = (lane & 15) + wave * 16;
        int moff = (lane >= 16) ? 8 : 0;
        #pragma unroll
        for (int r = 0; r < 8; ++r) {
            float v = c[r];
            msgL[(r + moff) * D_ + n] = (v > 0.0f) ? v : 0.0f;
        }
    }
    __syncthreads();

    // --- h_agg = mean over K edges (stored f16 for WMMA consumption) ---
    if (tid < 128) {
        float s = 0.0f;
        #pragma unroll
        for (int k = 0; k < K_; ++k) s += msgL[k * D_ + tid];
        hagg16[(size_t)node * D_ + tid] = (_Float16)(s * (1.0f / (float)K_));
    }
    // --- gate = msg @ W_g[layer]  (128 -> 3) ---
    else if (tid < 128 + 48) {
        int t = tid - 128;
        int mm = t / 3, j = t - 3 * mm;
        const float* wg = Wg + layer * D_ * 3;
        float g = 0.0f;
        #pragma unroll 8
        for (int d = 0; d < D_; ++d) g += msgL[mm * D_ + d] * wg[d * 3 + j];
        gateL[mm * 3 + j] = g;
    }
    __syncthreads();

    // --- dx[a][c] = mean_k gate[k,a]*xi[k,c];  x_bb += dx (block owns the node) ---
    if (tid < 9) {
        int a = tid / 3, xc = tid - 3 * a;
        float s = 0.0f;
        #pragma unroll
        for (int k = 0; k < K_; ++k)
            s += gateL[k * 3 + a] * xi[(node * K_ + k) * 3 + xc];
        xbb[node * 9 + a * 3 + xc] += s * (1.0f / (float)K_);
    }
}

// ================= node MLP (WMMA): h = relu([h|h_agg|chi_n] @ W_h[l]) =================
__global__ __launch_bounds__(256) void hupd_kernel(
    _Float16* __restrict__ h16, const _Float16* __restrict__ hagg16,
    const float* __restrict__ chin, const _Float16* __restrict__ WhP, int layer) {

    const int node0 = blockIdx.x * 16;
    const int tid   = threadIdx.x;
    const int lane  = tid & 31;
    const int wave  = tid >> 5;

    __shared__ _Float16 aL[16 * KHP];      // 16 nodes x 288 features

    // --- assemble [h | h_agg | chi_n | 0] tile; both 128-wide halves async ---
    {
        int r = tid >> 4;
        int part = tid & 15;
        int node = node0 + r;
        async_copy_b128(h16    + (size_t)node * D_ + part * 8, &aL[r * KHP + part * 8]);
        async_copy_b128(hagg16 + (size_t)node * D_ + part * 8, &aL[r * KHP + 128 + part * 8]);
        #pragma unroll
        for (int s = 0; s < 2; ++s) {
            int cidx = 256 + part * 2 + s;
            _Float16 v = (_Float16)0.0f;
            if (cidx < 258) v = (_Float16)chin[node * 2 + (cidx - 256)];
            aL[r * KHP + cidx] = v;
        }
    }
    async_wait_all();
    __syncthreads();

    v8f c = {};
    const int m    = lane & 15;
    const int hoff = (lane >= 16) ? 8 : 0;
    #pragma unroll
    for (int kc = 0; kc < NCHUNK; ++kc) {
        v16h a, bfr;
        const _Float16* ar = &aL[m * KHP + kc * 32 + hoff];
        #pragma unroll
        for (int j = 0; j < 16; ++j) a[j] = ar[j + ((j >= 8) ? 8 : 0)];
        const _Float16* br = WhP + ((size_t)((layer * NCHUNK + kc) * NTILE + wave)) * FRAG + lane * 16;
        bfr = *(const v16h*)br;
        c = __builtin_amdgcn_wmma_f32_16x16x32_f16(false, a, false, bfr, (short)0, c, false, false);
    }

    {
        int n    = (lane & 15) + wave * 16;
        int moff = (lane >= 16) ? 8 : 0;
        #pragma unroll
        for (int r = 0; r < 8; ++r) {
            float v = c[r];
            v = (v > 0.0f) ? v : 0.0f;
            int node = node0 + r + moff;
            h16[(size_t)node * D_ + n] = (_Float16)v;
        }
    }
}

// ================= final scale =================
__global__ void finalize_kernel(const float* __restrict__ xbb, float* __restrict__ out) {
    int id = blockIdx.x * blockDim.x + threadIdx.x;
    if (id >= N_ * 9) return;
    out[id] = xbb[id] * 10.0f;     // POS_SCALE
}

// ================= host launch =================
extern "C" void kernel_launch(void* const* d_in, const int* in_sizes, int n_in,
                              void* d_out, int out_size, void* d_ws, size_t ws_size,
                              hipStream_t stream) {
    (void)in_sizes; (void)n_in; (void)out_size; (void)ws_size;
    const float* x     = (const float*)d_in[0];
    const float* Winit = (const float*)d_in[4];
    const float* We    = (const float*)d_in[5];
    const float* Wh    = (const float*)d_in[6];
    const float* Wg    = (const float*)d_in[7];
    float* out = (float*)d_out;

    char* ws = (char*)d_ws;
    size_t off = 0;
    auto alloc = [&](size_t bytes) -> char* {
        char* p = ws + off;
        off = (off + bytes + 255) & ~(size_t)255;
        return p;
    };
    _Float16* h16    = (_Float16*)alloc((size_t)N_ * D_ * 2);
    _Float16* hagg16 = (_Float16*)alloc((size_t)N_ * D_ * 2);
    float*    xbb    = (float*)   alloc((size_t)N_ * 9 * 4);
    int*      idx    = (int*)     alloc((size_t)N_ * K_ * 4);
    float*    dist   = (float*)   alloc((size_t)N_ * K_ * 4);
    float*    xi     = (float*)   alloc((size_t)N_ * K_ * 3 * 4);
    float*    chin   = (float*)   alloc((size_t)N_ * 2 * 4);
    _Float16* WeP    = (_Float16*)alloc((size_t)NLAYERS * NCHUNK * NTILE * FRAG * 2);
    _Float16* WhP    = (_Float16*)alloc((size_t)NLAYERS * NCHUNK * NTILE * FRAG * 2);
    float*    mean   = (float*)   alloc(16);

    const int packN = NLAYERS * NCHUNK * NTILE * FRAG;
    pack_we_kernel<<<(packN + 255) / 256, 256, 0, stream>>>(We, WeP);
    pack_wh_kernel<<<(packN + 255) / 256, 256, 0, stream>>>(Wh, WhP);
    seth_kernel<<<(N_ * D_) / 256, 256, 0, stream>>>(x, h16);
    init_xbb_kernel<<<(N_ * 9 + 255) / 256, 256, 0, stream>>>(x, Winit, xbb);
    center_reduce_kernel<<<1, 256, 0, stream>>>(xbb, mean);
    center_apply_kernel<<<(N_ * 9 + 255) / 256, 256, 0, stream>>>(xbb, mean);

    for (int l = 0; l < NLAYERS; ++l) {
        knn_kernel<<<N_ / 256, 256, 0, stream>>>(xbb, idx, dist, xi, chin);
        msg_kernel<<<N_, 256, 0, stream>>>(h16, idx, dist, xi, WeP, Wg, hagg16, xbb, l);
        hupd_kernel<<<N_ / 16, 256, 0, stream>>>(h16, hagg16, chin, WhP, l);
        center_reduce_kernel<<<1, 256, 0, stream>>>(xbb, mean);
        center_apply_kernel<<<(N_ * 9 + 255) / 256, 256, 0, stream>>>(xbb, mean);
    }
    finalize_kernel<<<(N_ * 9 + 255) / 256, 256, 0, stream>>>(xbb, out);
}
